// MultiHeadAttention_57672820851426
// MI455X (gfx1250) — matrix operand order repaired and checked by reading, not compiled, not run on hardware
//
#include <hip/hip_runtime.h>
#include <cstdint>

// ---------- CDNA5 (gfx1250, wave32) types ----------
typedef __bf16 bf16_t;
typedef __attribute__((ext_vector_type(16))) __bf16  v16bf;
typedef __attribute__((ext_vector_type(8)))  __bf16  v8bf;
typedef __attribute__((ext_vector_type(4)))  __bf16  v4bf;
typedef __attribute__((ext_vector_type(8)))  float   v8f;
typedef __attribute__((ext_vector_type(4)))  uint32_t u32x4;
typedef __attribute__((ext_vector_type(8)))  uint32_t u32x8;

#define D_MODEL   1024
#define NUM_HEADS 16
#define DK        64
#define SEQ       2048
#define BATCH     4
#define MTOT      (BATCH * SEQ)   // 8192

// =====================================================================
// gfx1250 data-mover helpers
// =====================================================================

// Tensor Data Mover: 2D tile (tile_d0 x tile_d1 elems, 2-byte elems) from
// global (row stride stride0 elems) into LDS, padding each 128B row by 16B
// (D# pad_interval=32 DWORDs, pad_amount=4 DWORDs) to match [.][64+8] LDS rows.
// D# bitfields per CDNA5 ISA. Tracked: TENSORcnt.
__device__ inline void tdm_load_2d_to_lds(uint32_t lds_base, const void* gaddr,
                                          uint32_t tile_d0, uint32_t tile_d1,
                                          uint32_t stride0_elems)
{
    uint64_t ga = (uint64_t)(uintptr_t)gaddr;
    u32x4 g0;
    g0[0] = 1u;                                               // count=1 (user D#)
    g0[1] = lds_base;                                         // lds_addr (bytes)
    g0[2] = (uint32_t)ga;                                     // global_addr[31:0]
    g0[3] = (uint32_t)((ga >> 32) & 0x01FFFFFFu) | (2u << 30); // addr[56:32]|type=2
    u32x8 g1;
    g1[0] = (1u << 16)          // data_size = 2 bytes
          | (1u << 20)          // pad_enable
          | (4u << 22)          // pad_interval: 32 DWORDs (= one 128B row)
          | (3u << 25);         // pad_amount:   4 DWORDs (= 8 bf16)
    g1[1] = (tile_d0 & 0xFFFFu) << 16;                        // tensor_dim0 = tile_d0
    g1[2] = (tile_d0 >> 16) | ((tile_d1 & 0xFFFFu) << 16);    // tensor_dim1 = tile_d1
    g1[3] = (tile_d1 >> 16) | ((tile_d0 & 0xFFFFu) << 16);    // tile_dim0
    g1[4] = tile_d1 & 0xFFFFu;                                // tile_dim1 (tile_dim2=0)
    g1[5] = stride0_elems;                                    // tensor_dim0_stride lo32
    g1[6] = 0u;
    g1[7] = 0u;
    // 2 descriptor groups (<=2D tensor): VADDR2/VADDR3 = NULL
    asm volatile("tensor_load_to_lds %0, %1" :: "s"(g0), "s"(g1) : "memory");
}

// Async global -> LDS 16B copy (ASYNCcnt).
__device__ inline void async_ld_b128(uint32_t lds_addr, const void* src)
{
    asm volatile("global_load_async_to_lds_b128 %0, %1, off"
                 :: "v"(lds_addr), "v"(src) : "memory");
}
__device__ inline void wait_asynccnt0()
{
    asm volatile("s_wait_asynccnt 0" ::: "memory");
}

__device__ inline uint32_t lds_off(const void* p)
{
    return (uint32_t)(uintptr_t)p;   // generic LDS ptr: low 32 bits = LDS offset
}

// =====================================================================
// GEMM:  C[M,N] = A[M,K] @ W[N,K]^T + bias[N]
// Block tile 128x128, K-step 32, 8 waves: 4(M) x 2(N); each wave 32x64
// output = 2x4 tiles of v_wmma_f32_16x16x32_bf16 per k-step.
// =====================================================================
#define BM 128
#define BN 128
#define BK 32

template <typename AT, typename OT>
__global__ __launch_bounds__(256)
void gemm_bias_wmma(const AT* __restrict__ A, const float* __restrict__ W,
                    const float* __restrict__ bias, OT* __restrict__ C,
                    int M, int N, int K)
{
    // row stride (32+8)*2 = 80 B (multiple of 16 -> aligned v8bf chunks)
    __shared__ __align__(16) bf16_t As[BM][BK + 8];
    __shared__ __align__(16) bf16_t Ws[BN][BK + 8];

    const int tid  = threadIdx.x;
    const int lane = tid & 31;
    const int wave = tid >> 5;
    const int l15  = lane & 15;
    const int hl   = lane >> 4;

    const int m0 = blockIdx.x * BM;
    const int n0 = blockIdx.y * BN;
    const int wm = wave >> 1;         // 0..3 -> 32-row slice
    const int wn = wave & 1;          // 0..1 -> 64-col slice

    v8f acc[2][4] = {};

    for (int k0 = 0; k0 < K; k0 += BK) {
        // ---- stage A tile: 4 fp32 -> packed v4bf (8B ds_store) ----
        for (int idx = tid; idx < BM * (BK / 4); idx += 256) {
            int r = idx >> 3, c4 = (idx & 7) * 4;
            const AT* src = &A[(size_t)(m0 + r) * K + (k0 + c4)];
            v4bf pk;
            #pragma unroll
            for (int i = 0; i < 4; ++i) pk[i] = (bf16_t)(float)src[i];
            *reinterpret_cast<v4bf*>(&As[r][c4]) = pk;
        }
        // ---- stage W tile ----
        for (int idx = tid; idx < BN * (BK / 4); idx += 256) {
            int r = idx >> 3, c4 = (idx & 7) * 4;
            const float* src = &W[(size_t)(n0 + r) * K + (k0 + c4)];
            v4bf pk;
            #pragma unroll
            for (int i = 0; i < 4; ++i) pk[i] = (bf16_t)src[i];
            *reinterpret_cast<v4bf*>(&Ws[r][c4]) = pk;
        }
        // speculative prefetch of next A tile (global_prefetch_b8)
        if (k0 + BK < K)
            __builtin_prefetch(&A[(size_t)(m0 + (tid >> 1)) * K + (k0 + BK)], 0, 1);
        __syncthreads();

        // ---- A fragments: 16x32 bf16; lanes 0-15: K0-7/16-23, 16-31: K8-15/24-31
        v16bf af[2];
        #pragma unroll
        for (int mt = 0; mt < 2; ++mt) {
            const bf16_t* rowp = &As[wm * 32 + mt * 16 + l15][hl * 8];
            v8bf lo = *reinterpret_cast<const v8bf*>(rowp);
            v8bf hi = *reinterpret_cast<const v8bf*>(rowp + 16);
            #pragma unroll
            for (int i = 0; i < 8; ++i) { af[mt][i] = lo[i]; af[mt][i + 8] = hi[i]; }
        }
        // ---- B fragments: B = W^T (32x16); lane = N, lanes 0-15: K0-7/16-23, 16-31: K8-15/24-31
        v16bf wf_[4];
        #pragma unroll
        for (int nt = 0; nt < 4; ++nt) {
            const bf16_t* rowp = &Ws[wn * 64 + nt * 16 + l15][hl * 8];
            v8bf lo = *reinterpret_cast<const v8bf*>(rowp);
            v8bf hi = *reinterpret_cast<const v8bf*>(rowp + 16);
            #pragma unroll
            for (int i = 0; i < 8; ++i) { wf_[nt][i] = lo[i]; wf_[nt][i + 8] = hi[i]; }
        }
        #pragma unroll
        for (int mt = 0; mt < 2; ++mt)
            #pragma unroll
            for (int nt = 0; nt < 4; ++nt)
                acc[mt][nt] = __builtin_amdgcn_wmma_f32_16x16x32_bf16(
                    false, af[mt], false, wf_[nt], (short)0, acc[mt][nt], false, false);
        __syncthreads();
    }

    // ---- epilogue: C row = v + 8*hl, col = lane&15 ----
    #pragma unroll
    for (int mt = 0; mt < 2; ++mt)
        #pragma unroll
        for (int nt = 0; nt < 4; ++nt) {
            int col = n0 + wn * 64 + nt * 16 + l15;
            float bv = bias[col];
            #pragma unroll
            for (int v = 0; v < 8; ++v) {
                int row = m0 + wm * 32 + mt * 16 + v + hl * 8;
                C[(size_t)row * N + col] = (OT)(acc[mt][nt][v] + bv);
            }
        }
}

// =====================================================================
// Fused flash attention, one (b,h, 128-query block) per workgroup.
// K block staged by TDM (tensor_load_to_lds, TENSORcnt); V rows staged by
// global_load_async_to_lds_b128 (ASYNCcnt), then transposed LDS->LDS.
// scores & PV on v_wmma_f32_16x16x32_bf16; online softmax in registers.
// =====================================================================
#define BR 128
#define BC 128

__global__ __launch_bounds__(256)
void attention_wmma(const bf16_t* __restrict__ Q, const bf16_t* __restrict__ Km,
                    const bf16_t* __restrict__ V, bf16_t* __restrict__ O)
{
    __shared__ __align__(16) bf16_t Kb[BC][DK + 8];    // [key][dim], TDM dest
    __shared__ __align__(16) bf16_t Vr[BC][DK + 8];    // [key][dim], async dest
    __shared__ __align__(16) bf16_t Vt[DK][BC + 8];    // [dim][key]
    __shared__ __align__(16) bf16_t Ps[8][16][BC];     // per-wave P tile

    const int tid  = threadIdx.x;
    const int lane = tid & 31;
    const int wave = tid >> 5;
    const int l15  = lane & 15;
    const int hl   = lane >> 4;

    const int bh = blockIdx.y;
    const int b  = bh / NUM_HEADS;
    const int h  = bh % NUM_HEADS;
    const int q0 = blockIdx.x * BR + wave * 16;

    const size_t rowBase = (size_t)b * SEQ;
    const bf16_t* Qp = Q  + rowBase * D_MODEL + h * DK;
    const bf16_t* Kp = Km + rowBase * D_MODEL + h * DK;
    const bf16_t* Vp = V  + rowBase * D_MODEL + h * DK;

    // ---- Q fragments (A-layout 16x32, two frags cover d_k = 64) ----
    v16bf qf[2];
    {
        const bf16_t* qrow = Qp + (size_t)(q0 + l15) * D_MODEL;
        #pragma unroll
        for (int f = 0; f < 2; ++f) {
            const bf16_t* p = qrow + f * 32 + hl * 8;
            v8bf lo = *reinterpret_cast<const v8bf*>(p);
            v8bf hi = *reinterpret_cast<const v8bf*>(p + 16);
            #pragma unroll
            for (int i = 0; i < 8; ++i) { qf[f][i] = lo[i]; qf[f][i + 8] = hi[i]; }
        }
    }

    v8f   oacc[4] = {};
    float mrow[8], lrow[8];
    #pragma unroll
    for (int v = 0; v < 8; ++v) { mrow[v] = -1e30f; lrow[v] = 0.f; }

    for (int kv0 = 0; kv0 < SEQ; kv0 += BC) {
        __syncthreads();   // previous block's LDS reads complete

        // ---- K block via Tensor Data Mover (one wave issues the DMA) ----
        if (wave == 0)
            tdm_load_2d_to_lds(lds_off(&Kb[0][0]),
                               Kp + (size_t)kv0 * D_MODEL,
                               DK, BC, D_MODEL);

        // ---- V rows via async global->LDS copies (ASYNCcnt) ----
        for (int chunk = tid; chunk < BC * DK / 8; chunk += 256) {
            int key = chunk >> 3;
            int c8  = (chunk & 7) * 8;
            async_ld_b128(lds_off(&Vr[key][c8]),
                          Vp + (size_t)(kv0 + key) * D_MODEL + c8);
        }
        wait_asynccnt0();   // own wave's async copies landed in LDS

        // ---- transpose (each wave handles the chunks it loaded) ----
        for (int chunk = tid; chunk < BC * DK / 8; chunk += 256) {
            int key = chunk >> 3;
            int c8  = (chunk & 7) * 8;
            v8bf d = *reinterpret_cast<const v8bf*>(&Vr[key][c8]);
            #pragma unroll
            for (int i = 0; i < 8; ++i) Vt[c8 + i][key] = d[i];
        }

        if (wave == 0)
            __builtin_amdgcn_s_wait_tensorcnt(0);  // TDM tile complete
        __syncthreads();

        // ---- scores: 8 n-tiles of 16 keys, d_k = 64 in two WMMA steps ----
        float s[8][8];
        #pragma unroll
        for (int nt = 0; nt < 8; ++nt) {
            v8f sacc = {};
            #pragma unroll
            for (int f = 0; f < 2; ++f) {
                const bf16_t* kp = &Kb[nt * 16 + l15][f * 32 + hl * 8];
                v8bf lo = *reinterpret_cast<const v8bf*>(kp);
                v8bf hi = *reinterpret_cast<const v8bf*>(kp + 16);
                v16bf kf;
                #pragma unroll
                for (int i = 0; i < 8; ++i) { kf[i] = lo[i]; kf[i + 8] = hi[i]; }
                sacc = __builtin_amdgcn_wmma_f32_16x16x32_bf16(
                    false, qf[f], false, kf, (short)0, sacc, false, false);
            }
            #pragma unroll
            for (int v = 0; v < 8; ++v) s[nt][v] = sacc[v] * 0.125f;  // 1/sqrt(64)
        }

        // ---- online softmax (row = v + 8*hl; reduce over 16-lane group) ----
        float alpha[8];
        #pragma unroll
        for (int v = 0; v < 8; ++v) {
            float mx = s[0][v];
            #pragma unroll
            for (int nt = 1; nt < 8; ++nt) mx = fmaxf(mx, s[nt][v]);
            for (int msk = 1; msk < 16; msk <<= 1)
                mx = fmaxf(mx, __shfl_xor(mx, msk, 32));
            float mnew = fmaxf(mrow[v], mx);
            alpha[v]   = __expf(mrow[v] - mnew);
            mrow[v]    = mnew;
            float rsum = 0.f;
            #pragma unroll
            for (int nt = 0; nt < 8; ++nt) {
                float p = __expf(s[nt][v] - mnew);
                s[nt][v] = p;
                rsum += p;
            }
            for (int msk = 1; msk < 16; msk <<= 1)
                rsum += __shfl_xor(rsum, msk, 32);
            lrow[v] = lrow[v] * alpha[v] + rsum;
        }
        #pragma unroll
        for (int dt = 0; dt < 4; ++dt)
            #pragma unroll
            for (int v = 0; v < 8; ++v) oacc[dt][v] *= alpha[v];

        // ---- C-layout -> A-layout for P via per-wave LDS scratch ----
        #pragma unroll
        for (int nt = 0; nt < 8; ++nt)
            #pragma unroll
            for (int v = 0; v < 8; ++v)
                Ps[wave][v + hl * 8][nt * 16 + l15] = (bf16_t)s[nt][v];

        v16bf pf[4];
        #pragma unroll
        for (int kt = 0; kt < 4; ++kt) {
            const bf16_t* p = &Ps[wave][l15][kt * 32 + hl * 8];
            v8bf lo = *reinterpret_cast<const v8bf*>(p);
            v8bf hi = *reinterpret_cast<const v8bf*>(p + 16);
            #pragma unroll
            for (int i = 0; i < 8; ++i) { pf[kt][i] = lo[i]; pf[kt][i + 8] = hi[i]; }
        }

        // ---- O += P @ V ----
        #pragma unroll
        for (int dt = 0; dt < 4; ++dt) {
            #pragma unroll
            for (int kt = 0; kt < 4; ++kt) {
                const bf16_t* vp = &Vt[dt * 16 + l15][kt * 32 + hl * 8];
                v8bf lo = *reinterpret_cast<const v8bf*>(vp);
                v8bf hi = *reinterpret_cast<const v8bf*>(vp + 16);
                v16bf vf;
                #pragma unroll
                for (int i = 0; i < 8; ++i) { vf[i] = lo[i]; vf[i + 8] = hi[i]; }
                oacc[dt] = __builtin_amdgcn_wmma_f32_16x16x32_bf16(
                    false, pf[kt], false, vf, (short)0, oacc[dt], false, false);
            }
        }
    }

    // ---- normalize and store (bf16, combined-heads layout [B*S, D]) ----
    float inv[8];
    #pragma unroll
    for (int v = 0; v < 8; ++v) inv[v] = 1.f / lrow[v];
    #pragma unroll
    for (int dt = 0; dt < 4; ++dt)
        #pragma unroll
        for (int v = 0; v < 8; ++v) {
            int qr  = q0 + v + hl * 8;
            int dim = dt * 16 + l15;
            O[(rowBase + qr) * D_MODEL + h * DK + dim] = (bf16_t)(oacc[dt][v] * inv[v]);
        }
}

// =====================================================================
// Launcher
// =====================================================================
extern "C" void kernel_launch(void* const* d_in, const int* in_sizes, int n_in,
                              void* d_out, int out_size, void* d_ws, size_t ws_size,
                              hipStream_t stream) {
    const float* query = (const float*)d_in[0];
    const float* key   = (const float*)d_in[1];
    const float* value = (const float*)d_in[2];
    const float* W_q   = (const float*)d_in[3];
    const float* b_q   = (const float*)d_in[4];
    const float* W_k   = (const float*)d_in[5];
    const float* b_k   = (const float*)d_in[6];
    const float* W_v   = (const float*)d_in[7];
    const float* b_v   = (const float*)d_in[8];
    const float* W_out = (const float*)d_in[9];
    const float* b_out = (const float*)d_in[10];
    float* out = (float*)d_out;

    const size_t elems = (size_t)MTOT * D_MODEL;
    bf16_t* qb = (bf16_t*)d_ws;         // 16 MB each
    bf16_t* kb = qb + elems;
    bf16_t* vb = kb + elems;
    bf16_t* ab = vb + elems;

    dim3 gemm_grid(MTOT / BM, D_MODEL / BN);   // (64, 8)
    gemm_bias_wmma<float, bf16_t><<<gemm_grid, 256, 0, stream>>>(
        query, W_q, b_q, qb, MTOT, D_MODEL, D_MODEL);
    gemm_bias_wmma<float, bf16_t><<<gemm_grid, 256, 0, stream>>>(
        key, W_k, b_k, kb, MTOT, D_MODEL, D_MODEL);
    gemm_bias_wmma<float, bf16_t><<<gemm_grid, 256, 0, stream>>>(
        value, W_v, b_v, vb, MTOT, D_MODEL, D_MODEL);

    dim3 att_grid(SEQ / BR, BATCH * NUM_HEADS);  // (16, 64)
    attention_wmma<<<att_grid, 256, 0, stream>>>(qb, kb, vb, ab);

    gemm_bias_wmma<bf16_t, float><<<gemm_grid, 256, 0, stream>>>(
        ab, W_out, b_out, out, MTOT, D_MODEL, D_MODEL);
}
